// Categ_CrisCasTransformer_833223656494
// MI455X (gfx1250) — matrix operand order repaired and checked by reading, not compiled, not run on hardware
//
#include <hip/hip_runtime.h>
#include <hip/hip_bf16.h>
#include <cstdint>

#define B_  32768
#define S_  20
#define D_  64
#define L_  12
#define H_  8
#define C_  2
#define FF_ 128

// f16 weight workspace layout (element offsets)
#define OFF_WQ 0
#define OFF_WK 393216
#define OFF_WV 786432
#define OFF_WZ 1179648
#define OFF_W1 1572864
#define OFF_W2 1671168
#define W_TOTAL 1769472

typedef _Float16 v16h __attribute__((ext_vector_type(16)));
typedef _Float16 v8h  __attribute__((ext_vector_type(8)));
typedef _Float16 v4h  __attribute__((ext_vector_type(4)));
typedef _Float16 v2h  __attribute__((ext_vector_type(2)));
typedef float    v8f  __attribute__((ext_vector_type(8)));

// ---------------------------------------------------------------------------
// Prologue: convert + transpose all GEMM weights to f16 [N][K] row-major.
// ---------------------------------------------------------------------------
__global__ void convert_weights(const float* __restrict__ Wq,
                                const float* __restrict__ Wk,
                                const float* __restrict__ Wv,
                                const float* __restrict__ Wz,
                                const float* __restrict__ w1,
                                const float* __restrict__ w2,
                                _Float16* __restrict__ W16) {
  int i = blockIdx.x * blockDim.x + threadIdx.x;
  if (i >= W_TOTAL) return;
  float val;
  if (i < OFF_WK) {                       // WqT[l,h,e,d] = Wq[l,h,d,e]
    int j = i - OFF_WQ;
    int l = j / 32768, r = j % 32768, h = r / 4096, rr = r % 4096;
    int e = rr / 64, d = rr % 64;
    val = Wq[(((l * H_ + h) * 64 + d) * 64) + e];
  } else if (i < OFF_WV) {
    int j = i - OFF_WK;
    int l = j / 32768, r = j % 32768, h = r / 4096, rr = r % 4096;
    int e = rr / 64, d = rr % 64;
    val = Wk[(((l * H_ + h) * 64 + d) * 64) + e];
  } else if (i < OFF_WZ) {
    int j = i - OFF_WV;
    int l = j / 32768, r = j % 32768, h = r / 4096, rr = r % 4096;
    int e = rr / 64, d = rr % 64;
    val = Wv[(((l * H_ + h) * 64 + d) * 64) + e];
  } else if (i < OFF_W1) {                // WzT[l,d',k] = Wz[l,k,d']
    int j = i - OFF_WZ;
    int l = j / 32768, r = j % 32768, dp = r / 512, k = r % 512;
    val = Wz[(l * 512 + k) * 64 + dp];
  } else if (i < OFF_W2) {                // w1T[l,f,d] = w1[l,d,f]
    int j = i - OFF_W1;
    int l = j / 8192, r = j % 8192, f = r / 64, d = r % 64;
    val = w1[(l * 64 + d) * 128 + f];
  } else {                                // w2T[l,d,f] = w2[l,f,d]
    int j = i - OFF_W2;
    int l = j / 8192, r = j % 8192, d = r / 128, f = r % 128;
    val = w2[(l * 128 + f) * 64 + d];
  }
  W16[i] = (_Float16)val;
}

// ---------------------------------------------------------------------------
// WMMA fragment helpers (wave32 layouts per CDNA5 ISA 7.12.2)
// ---------------------------------------------------------------------------
__device__ __forceinline__ v8f wmma16(v16h a, v16h b, v8f c) {
  return __builtin_amdgcn_wmma_f32_16x16x32_f16(false, a, false, b,
                                                (short)0, c, false, false);
}
// f16-accumulating variant for intermediates (q/k/v/z/hidden)
__device__ __forceinline__ v8h wmma16h(v16h a, v16h b, v8h c) {
  return __builtin_amdgcn_wmma_f16_16x16x32_f16(false, a, false, b,
                                                (short)0, c, false, false);
}

// A-fragment 16x32 f16 from row-major [M][K] (ld in halves).
__device__ __forceinline__ v16h ldA16(const _Float16* base, int ld, int mrow,
                                      int kb, int lane) {
  int r = lane & 15, hi = lane >> 4;
  const _Float16* p = base + (mrow + r) * ld + kb + hi * 8;
  v8h lo = *(const v8h*)p;
  v8h hv = *(const v8h*)(p + 16);
  v16h a;
#pragma unroll
  for (int i = 0; i < 8; i++) { a[i] = lo[i]; a[i + 8] = hv[i]; }
  return a;
}

// B-fragment 32x16 f16 from [N][K] row-major (pre-transposed weights / LDS).
__device__ __forceinline__ v16h ldB16(const _Float16* base, int ld, int ncol,
                                      int kb, int lane) {
  int n = lane & 15, hi = lane >> 4;
  const _Float16* p = base + (ncol + n) * ld + kb + hi * 8;
  v8h lo = *(const v8h*)p;
  v8h hv = *(const v8h*)(p + 16);
  v16h b;
#pragma unroll
  for (int i = 0; i < 8; i++) { b[i] = lo[i]; b[i + 8] = hv[i]; }
  return b;
}

// f32 C/D tile: VGPR j -> row mrow + (lane>=16)*8 + j, col ncol + (lane&15)
__device__ __forceinline__ void stC_f(float* base, int ld, int mrow, int ncol,
                                      v8f c, int lane) {
  int r0 = mrow + ((lane >> 4) << 3), col = ncol + (lane & 15);
#pragma unroll
  for (int j = 0; j < 8; j++) base[(r0 + j) * ld + col] = c[j];
}
// f16 C/D tile (8 halves): half j -> row mrow + (lane>=16)*8 + j, col ncol+(lane&15)
__device__ __forceinline__ void stD_h(_Float16* base, int ld, int mrow,
                                      int ncol, v8h c, int lane) {
  int r0 = mrow + ((lane >> 4) << 3), col = ncol + (lane & 15);
#pragma unroll
  for (int j = 0; j < 8; j++) base[(r0 + j) * ld + col] = c[j];
}
__device__ __forceinline__ void stD_hT(_Float16* base, int ld, int mrow,
                                       int ncol, v8h c, int lane) {  // transposed
  int r0 = mrow + ((lane >> 4) << 3), col = ncol + (lane & 15);
#pragma unroll
  for (int j = 0; j < 8; j++) base[col * ld + (r0 + j)] = c[j];
}

// ---- vectorized row utilities ------------------------------------------------
__device__ __forceinline__ void stage_f16_row(_Float16* dst, const float* src) {
#pragma unroll
  for (int g = 0; g < 16; g++) {
    float4 x4 = ((const float4*)src)[g];
    v4h h;
    h[0] = (_Float16)x4.x; h[1] = (_Float16)x4.y;
    h[2] = (_Float16)x4.z; h[3] = (_Float16)x4.w;
    *(v4h*)(dst + g * 4) = h;
  }
}
__device__ __forceinline__ void zero_f32_row(float* dst) {
  float4 z4 = make_float4(0.f, 0.f, 0.f, 0.f);
#pragma unroll
  for (int g = 0; g < 16; g++) ((float4*)dst)[g] = z4;
}

// residual + LayerNorm over one 64-wide row, in place in xrow.
template <bool BIAS>
__device__ __forceinline__ void resid_ln_row(float* xrow, const float* orow,
                                             const float* bias, const float* g,
                                             const float* bvec) {
  float sum = 0.f, sq = 0.f;
  float4* xr4 = (float4*)xrow;
  const float4* or4 = (const float4*)orow;
#pragma unroll
  for (int i = 0; i < 16; i++) {
    float4 y4 = or4[i];
    float4 x4 = xr4[i];
    if (BIAS) {
      float4 b4 = ((const float4*)bias)[i];
      y4.x += b4.x; y4.y += b4.y; y4.z += b4.z; y4.w += b4.w;
    }
    y4.x += x4.x; y4.y += x4.y; y4.z += x4.z; y4.w += x4.w;
    xr4[i] = y4;
    sum += y4.x + y4.y + y4.z + y4.w;
    sq += y4.x * y4.x + y4.y * y4.y + y4.z * y4.z + y4.w * y4.w;
  }
  float mean = sum * (1.f / 64.f);
  float var = sq * (1.f / 64.f) - mean * mean;
  float r = rsqrtf(var + 1e-5f);
  const float4* g4 = (const float4*)g;
  const float4* b4p = (const float4*)bvec;
#pragma unroll
  for (int i = 0; i < 16; i++) {
    float4 x4 = xr4[i];
    float4 gg = g4[i], bb = b4p[i];
    x4.x = (x4.x - mean) * r * gg.x + bb.x;
    x4.y = (x4.y - mean) * r * gg.y + bb.y;
    x4.z = (x4.z - mean) * r * gg.z + bb.z;
    x4.w = (x4.w - mean) * r * gg.w + bb.w;
    xr4[i] = x4;
  }
}

#define WAVE_LDS 36864
#define WAVES_PER_WG 4

// ---------------------------------------------------------------------------
// Fused 12-layer transformer: 1 wave = 1 batch element, all layers in LDS.
// ---------------------------------------------------------------------------
__global__ __launch_bounds__(128) void transformer_fused(
    const int* __restrict__ X, const float* __restrict__ nuc_emb,
    const float* __restrict__ pos_emb, const float* __restrict__ Wz_b,
    const float* __restrict__ ln1_g, const float* __restrict__ ln1_b,
    const float* __restrict__ b1, const float* __restrict__ b2,
    const float* __restrict__ ln2_g, const float* __restrict__ ln2_b,
    const float* __restrict__ queryv, const float* __restrict__ Wy_w,
    const float* __restrict__ Wy_b, const _Float16* __restrict__ W16,
    float* __restrict__ out) {
  extern __shared__ char smem[];
  const int lane = threadIdx.x & 31;
  const int wid = threadIdx.x >> 5;
  const int b = blockIdx.x * WAVES_PER_WG + wid;

  char* wb = smem + wid * WAVE_LDS;
  float*    xs = (float*)(wb);                 // [32][64] f32 activations
  _Float16* aX = (_Float16*)(wb + 8192);       // [32][64] f16 A staging (x / y1)
  _Float16* aQ = (_Float16*)(wb + 12288);      // [32][64] f16 q, then [32][32] probs
  _Float16* kb = (_Float16*)(wb + 16384);      // [32][64] f16 k, then z
  _Float16* vb = (_Float16*)(wb + 20480);      // [64][32] f16 v^T
  float*    sb = (float*)(wb + 24576);         // [32][32] f32 scores / pool scratch
  float*    hbF = (float*)(wb + 28672);        // 8KB: o / y2 dump (f32 [32][64])
  _Float16* hbH = (_Float16*)(wb + 28672);     //      or hidden (f16 [32][128])

  // ---- embedding: x = nuc_emb[X] + pos_emb, rows >= 20 zeroed ----
  if (lane < S_) {
    int tok = X[b * S_ + lane];
    const float4* nr = (const float4*)(nuc_emb + tok * D_);
    const float4* pr = (const float4*)(pos_emb + lane * D_);
    float4* xr = (float4*)(xs + lane * D_);
#pragma unroll
    for (int g = 0; g < 16; g++) {
      float4 n4 = nr[g], p4 = pr[g];
      xr[g] = make_float4(n4.x + p4.x, n4.y + p4.y, n4.z + p4.z, n4.w + p4.w);
    }
  } else {
    zero_f32_row(xs + lane * D_);
  }

  const unsigned long long attn_base =
      (unsigned long long)B_ * C_ + (unsigned long long)B_ * S_;

  for (int l = 0; l < L_; l++) {
    // stage x -> f16
    stage_f16_row(aX + lane * D_, xs + lane * D_);
    asm volatile("s_wait_dscnt 0x0" ::: "memory");

    // preload x A-fragments; invariant across all 8 heads
    v16h ax[2][2];
#pragma unroll
    for (int m = 0; m < 2; m++)
#pragma unroll
      for (int kk = 0; kk < 2; kk++)
        ax[m][kk] = ldA16(aX, 64, m * 16, kk * 32, lane);

    v8f o[2][4];
#pragma unroll
    for (int m = 0; m < 2; m++)
#pragma unroll
      for (int n = 0; n < 4; n++) o[m][n] = (v8f){};

    for (int h = 0; h < H_; h++) {
      const _Float16* WqT = W16 + OFF_WQ + (l * H_ + h) * 4096;
      const _Float16* WkT = W16 + OFF_WK + (l * H_ + h) * 4096;
      const _Float16* WvT = W16 + OFF_WV + (l * H_ + h) * 4096;
      __builtin_prefetch(WqT + 4096, 0, 1);

      // ---- q/k/v projections: f16 accumulate, one B load feeds both m-tiles ----
#pragma unroll
      for (int n = 0; n < 4; n++) {
        v8h aq[2] = {(v8h){}, (v8h){}};
        v8h ak[2] = {(v8h){}, (v8h){}};
        v8h av[2] = {(v8h){}, (v8h){}};
#pragma unroll
        for (int kk = 0; kk < 2; kk++) {
          v16h bq = ldB16(WqT, 64, n * 16, kk * 32, lane);
          v16h bk = ldB16(WkT, 64, n * 16, kk * 32, lane);
          v16h bv = ldB16(WvT, 64, n * 16, kk * 32, lane);
#pragma unroll
          for (int m = 0; m < 2; m++) {
            aq[m] = wmma16h(ax[m][kk], bq, aq[m]);
            ak[m] = wmma16h(ax[m][kk], bk, ak[m]);
            av[m] = wmma16h(ax[m][kk], bv, av[m]);
          }
        }
#pragma unroll
        for (int m = 0; m < 2; m++) {
          stD_h(aQ, 64, m * 16, n * 16, aq[m], lane);
          stD_h(kb, 64, m * 16, n * 16, ak[m], lane);
          stD_hT(vb, 32, m * 16, n * 16, av[m], lane);  // v^T[e][t]
        }
      }
      asm volatile("s_wait_dscnt 0x0" ::: "memory");

      // ---- scores = q @ k^T  (K = e = 64), f32 accumulate ----
      v16h qf[2][2];
#pragma unroll
      for (int m = 0; m < 2; m++)
#pragma unroll
        for (int kk = 0; kk < 2; kk++)
          qf[m][kk] = ldA16(aQ, 64, m * 16, kk * 32, lane);
#pragma unroll
      for (int n = 0; n < 2; n++) {
        v8f s8[2] = {(v8f){}, (v8f){}};
#pragma unroll
        for (int kk = 0; kk < 2; kk++) {
          v16h bk = ldB16(kb, 64, n * 16, kk * 32, lane);
#pragma unroll
          for (int m = 0; m < 2; m++) s8[m] = wmma16(qf[m][kk], bk, s8[m]);
        }
#pragma unroll
        for (int m = 0; m < 2; m++) stC_f(sb, 32, m * 16, n * 16, s8[m], lane);
      }
      asm volatile("s_wait_dscnt 0x0" ::: "memory");

      // ---- softmax (lane = row s); write attn map; probs f16 into aQ ----
      if (lane < S_) {
        float v[S_];
        const float4* sr = (const float4*)(sb + lane * 32);
#pragma unroll
        for (int g = 0; g < 5; g++) {
          float4 t4 = sr[g];
          v[g * 4 + 0] = t4.x; v[g * 4 + 1] = t4.y;
          v[g * 4 + 2] = t4.z; v[g * 4 + 3] = t4.w;
        }
        float mx = -1e30f;
#pragma unroll
        for (int t = 0; t < S_; t++) mx = fmaxf(mx, v[t]);
        float sum = 0.f;
#pragma unroll
        for (int t = 0; t < S_; t++) {
          float e = __expf((v[t] - mx) * 0.125f);
          v[t] = e; sum += e;
        }
        float inv = 1.f / sum;
#pragma unroll
        for (int t = 0; t < S_; t++) v[t] *= inv;
        unsigned long long row =
            attn_base +
            ((((unsigned long long)l * B_ + b) * H_ + h) * S_ + lane) * S_;
        float4* op = (float4*)(out + row);
#pragma unroll
        for (int q4 = 0; q4 < 5; q4++)
          op[q4] = make_float4(v[q4 * 4], v[q4 * 4 + 1], v[q4 * 4 + 2],
                               v[q4 * 4 + 3]);
        _Float16* pr = aQ + lane * 32;
#pragma unroll
        for (int t = 0; t < 10; t++) {
          v2h h2; h2[0] = (_Float16)v[2 * t]; h2[1] = (_Float16)v[2 * t + 1];
          *(v2h*)(pr + 2 * t) = h2;
        }
        *(v4h*)(pr + 20) = (v4h){};
        *(v8h*)(pr + 24) = (v8h){};
      } else {
        _Float16* pr = aQ + lane * 32;
#pragma unroll
        for (int g = 0; g < 4; g++) *(v8h*)(pr + g * 8) = (v8h){};
      }
      asm volatile("s_wait_dscnt 0x0" ::: "memory");

      // ---- z = probs @ v  (K = t = 32, one step, f16 accumulate) -> kb ----
      v16h pf[2];
#pragma unroll
      for (int m = 0; m < 2; m++) pf[m] = ldA16(aQ, 32, m * 16, 0, lane);
#pragma unroll
      for (int n = 0; n < 4; n++) {
        v16h bv = ldB16(vb, 32, n * 16, 0, lane);
#pragma unroll
        for (int m = 0; m < 2; m++) {
          v8h z8 = wmma16h(pf[m], bv, (v8h){});
          stD_h(kb, 64, m * 16, n * 16, z8, lane);
        }
      }
      asm volatile("s_wait_dscnt 0x0" ::: "memory");

      // ---- output projection slice, accumulate across heads (f32) ----
      const _Float16* WzT = W16 + OFF_WZ + l * 64 * 512;
      v16h zf[2][2];
#pragma unroll
      for (int m = 0; m < 2; m++)
#pragma unroll
        for (int kk = 0; kk < 2; kk++)
          zf[m][kk] = ldA16(kb, 64, m * 16, kk * 32, lane);
#pragma unroll
      for (int n = 0; n < 4; n++)
#pragma unroll
        for (int kk = 0; kk < 2; kk++) {
          v16h bz = ldB16(WzT, 512, n * 16, h * 64 + kk * 32, lane);
#pragma unroll
          for (int m = 0; m < 2; m++) o[m][n] = wmma16(zf[m][kk], bz, o[m][n]);
        }
    }  // heads

    // ---- residual + LN1 -> y1 (into xs) ----
#pragma unroll
    for (int m = 0; m < 2; m++)
#pragma unroll
      for (int n = 0; n < 4; n++) stC_f(hbF, 64, m * 16, n * 16, o[m][n], lane);
    asm volatile("s_wait_dscnt 0x0" ::: "memory");
    if (lane < S_) {
      resid_ln_row<true>(xs + lane * 64, hbF + lane * 64, Wz_b + l * 64,
                         ln1_g + l * 64, ln1_b + l * 64);
    } else {
      zero_f32_row(xs + lane * 64);
    }

    // re-stage y1 -> f16
    stage_f16_row(aX + lane * D_, xs + lane * D_);
    asm volatile("s_wait_dscnt 0x0" ::: "memory");

    // ---- MLP GEMM1 (64 -> 128) + bias + relu -> hidden f16 (f16 accumulate) ----
    const _Float16* w1T = W16 + OFF_W1 + l * FF_ * D_;
    const _Float16* w2T = W16 + OFF_W2 + l * D_ * FF_;
    v16h yf[2][2];
#pragma unroll
    for (int m = 0; m < 2; m++)
#pragma unroll
      for (int kk = 0; kk < 2; kk++)
        yf[m][kk] = ldA16(aX, 64, m * 16, kk * 32, lane);
#pragma unroll
    for (int n = 0; n < 8; n++) {
      _Float16 bias = (_Float16)b1[l * FF_ + n * 16 + (lane & 15)];
      v8h h8[2];
#pragma unroll
      for (int m = 0; m < 2; m++)
#pragma unroll
        for (int j = 0; j < 8; j++) h8[m][j] = bias;
#pragma unroll
      for (int kk = 0; kk < 2; kk++) {
        v16h bw = ldB16(w1T, 64, n * 16, kk * 32, lane);
#pragma unroll
        for (int m = 0; m < 2; m++) h8[m] = wmma16h(yf[m][kk], bw, h8[m]);
      }
#pragma unroll
      for (int m = 0; m < 2; m++) {
#pragma unroll
        for (int j = 0; j < 8; j++)
          h8[m][j] = h8[m][j] > (_Float16)0.f ? h8[m][j] : (_Float16)0.f;
        stD_h(hbH, 128, m * 16, n * 16, h8[m], lane);
      }
    }
    asm volatile("s_wait_dscnt 0x0" ::: "memory");

    // ---- MLP GEMM2 (128 -> 64) + bias, f32 accumulate; hold tiles, dump ----
    v16h hf[2][4];
#pragma unroll
    for (int m = 0; m < 2; m++)
#pragma unroll
      for (int kk = 0; kk < 4; kk++)
        hf[m][kk] = ldA16(hbH, 128, m * 16, kk * 32, lane);
    v8f y2[2][4];
#pragma unroll
    for (int m = 0; m < 2; m++)
#pragma unroll
      for (int n = 0; n < 4; n++) {
        float bias = b2[l * 64 + n * 16 + (lane & 15)];
#pragma unroll
        for (int j = 0; j < 8; j++) y2[m][n][j] = bias;
      }
#pragma unroll
    for (int n = 0; n < 4; n++)
#pragma unroll
      for (int kk = 0; kk < 4; kk++) {
        v16h bw = ldB16(w2T, 128, n * 16, kk * 32, lane);
#pragma unroll
        for (int m = 0; m < 2; m++) y2[m][n] = wmma16(hf[m][kk], bw, y2[m][n]);
      }
#pragma unroll
    for (int m = 0; m < 2; m++)
#pragma unroll
      for (int n = 0; n < 4; n++) stC_f(hbF, 64, m * 16, n * 16, y2[m][n], lane);
    asm volatile("s_wait_dscnt 0x0" ::: "memory");

    // ---- residual + LN2 -> new x ----
    if (lane < S_) {
      resid_ln_row<false>(xs + lane * 64, hbF + lane * 64, nullptr,
                          ln2_g + l * 64, ln2_b + l * 64);
    } else {
      zero_f32_row(xs + lane * 64);
    }
    asm volatile("s_wait_dscnt 0x0" ::: "memory");
  }  // layers

  // ---- attention pooling head ----
  if (lane < S_) {
    float dot = 0.f;
    const float4* xr = (const float4*)(xs + lane * 64);
    const float4* qr = (const float4*)queryv;
#pragma unroll
    for (int g = 0; g < 16; g++) {
      float4 x4 = xr[g], q4 = qr[g];
      dot += x4.x * q4.x + x4.y * q4.y + x4.z * q4.z + x4.w * q4.w;
    }
    sb[lane] = dot * 0.125f;  // /SCALE^2
  }
  asm volatile("s_wait_dscnt 0x0" ::: "memory");
  float mx = -1e30f;
  for (int t = 0; t < S_; t++) mx = fmaxf(mx, sb[t]);
  float sum = 0.f;
  for (int t = 0; t < S_; t++) sum += __expf(sb[t] - mx);
  float inv = 1.f / sum;
  if (lane < S_) {
    float pw = __expf(sb[lane] - mx) * inv;
    out[(unsigned long long)B_ * C_ + (unsigned long long)b * S_ + lane] = pw;
    sb[32 + lane] = pw;
  }
  asm volatile("s_wait_dscnt 0x0" ::: "memory");
  // pooled: lane handles dims lane and lane+32
  float p0 = 0.f, p1 = 0.f;
  for (int s = 0; s < S_; s++) {
    float pw = sb[32 + s];
    p0 += pw * xs[s * 64 + lane];
    p1 += pw * xs[s * 64 + 32 + lane];
  }
  sb[64 + lane] = p0;
  sb[96 + lane] = p1;
  asm volatile("s_wait_dscnt 0x0" ::: "memory");
  if (lane == 0) {
    float l0 = Wy_b[0], l1 = Wy_b[1];
    for (int d = 0; d < D_; d++) {
      float pd = sb[64 + d];
      l0 += pd * Wy_w[d * 2 + 0];
      l1 += pd * Wy_w[d * 2 + 1];
    }
    float m2 = fmaxf(l0, l1);
    float lse = m2 + __logf(__expf(l0 - m2) + __expf(l1 - m2));
    out[(unsigned long long)b * 2 + 0] = l0 - lse;
    out[(unsigned long long)b * 2 + 1] = l1 - lse;
  }
}

// ---------------------------------------------------------------------------
extern "C" void kernel_launch(void* const* d_in, const int* in_sizes, int n_in,
                              void* d_out, int out_size, void* d_ws,
                              size_t ws_size, hipStream_t stream) {
  const int* X = (const int*)d_in[0];
  const float* nuc_emb = (const float*)d_in[1];
  const float* pos_emb = (const float*)d_in[2];
  const float* Wq = (const float*)d_in[3];
  const float* Wk = (const float*)d_in[4];
  const float* Wv = (const float*)d_in[5];
  const float* Wz_w = (const float*)d_in[6];
  const float* Wz_b = (const float*)d_in[7];
  const float* ln1_g = (const float*)d_in[8];
  const float* ln1_b = (const float*)d_in[9];
  const float* w1 = (const float*)d_in[10];
  const float* b1 = (const float*)d_in[11];
  const float* w2 = (const float*)d_in[12];
  const float* b2 = (const float*)d_in[13];
  const float* ln2_g = (const float*)d_in[14];
  const float* ln2_b = (const float*)d_in[15];
  const float* queryv = (const float*)d_in[16];
  const float* Wy_w = (const float*)d_in[17];
  const float* Wy_b = (const float*)d_in[18];
  float* out = (float*)d_out;

  _Float16* W16 = (_Float16*)d_ws;

  convert_weights<<<(W_TOTAL + 255) / 256, 256, 0, stream>>>(Wq, Wk, Wv, Wz_w,
                                                             w1, w2, W16);

  transformer_fused<<<B_ / WAVES_PER_WG, 32 * WAVES_PER_WG,
                      WAVES_PER_WG * WAVE_LDS, stream>>>(
      X, nuc_emb, pos_emb, Wz_b, ln1_g, ln1_b, b1, b2, ln2_g, ln2_b, queryv,
      Wy_w, Wy_b, W16, out);
}
